// GCNout_26310969655756
// MI455X (gfx1250) — compile-verified, matrix operand order checked by reading, not run on hardware
//
#include <hip/hip_runtime.h>
#include <math.h>

// ---------------------------------------------------------------------------
// GCN gated-relation aggregation for MI455X (gfx1250, wave32, WMMA).
//
// Pipeline (all on `stream`):
//   1. zero_kernel      : zero nb accumulators (ws) + u_emb slice of d_out
//   2. spmm_kernel x4   : edge-parallel scatter-add (1 edge per wave,
//                         float4/lane gather, global_atomic_add_f32 scatter)
//   3. gate_kernel      : fused (x*nb) -> f16 WMMA vs 128x128 weights ->
//                         leaky_relu -> rowsum -> softmax(3) -> blended
//                         neighbor output. 16-row tile per wave.
// ---------------------------------------------------------------------------

typedef _Float16 v4h  __attribute__((ext_vector_type(4)));
typedef _Float16 v8h  __attribute__((ext_vector_type(8)));
typedef _Float16 v16h __attribute__((ext_vector_type(16)));
typedef float    v8f  __attribute__((ext_vector_type(8)));

#define DIM 128

__global__ void zero_kernel(float* __restrict__ p, size_t n4) {
  size_t i = (size_t)blockIdx.x * blockDim.x + threadIdx.x;
  size_t stride = (size_t)gridDim.x * blockDim.x;
  const float4 z = make_float4(0.f, 0.f, 0.f, 0.f);
  for (; i < n4; i += stride) ((float4*)p)[i] = z;
}

// out[dst[e], :] += val[e] * x[src[e], :]   (one edge per wave32)
__global__ void spmm_kernel(const int* __restrict__ src,
                            const int* __restrict__ dst,
                            const float* __restrict__ val,
                            const float* __restrict__ x,
                            float* __restrict__ out, int nE) {
  const int gtid = blockIdx.x * blockDim.x + threadIdx.x;
  const int e = gtid >> 5;           // one wave32 per edge
  const int lane = threadIdx.x & 31; // 32 lanes x float4 = 128 floats
  if (e >= nE) return;
  const int s = src[e];
  const int d = dst[e];
  const float w = val[e];
  const float4 v = ((const float4*)(x + (size_t)s * DIM))[lane];
  float* o = out + (size_t)d * DIM + lane * 4;
  unsafeAtomicAdd(o + 0, v.x * w);
  unsafeAtomicAdd(o + 1, v.y * w);
  unsafeAtomicAdd(o + 2, v.z * w);
  unsafeAtomicAdd(o + 3, v.w * w);
}

// Fused gating: per 16-row tile (one per wave):
//   for rel: a = f16(x*nb_rel); h = a @ W_rel (WMMA); s_rel = rowsum(leaky(h))/sqrt(D)
//   score = softmax(s); neighbor = sum_rel score_rel * nb_rel
__launch_bounds__(128, 1)
__global__ void gate_kernel(const float* __restrict__ x,
                            const float* __restrict__ nb0,
                            const float* __restrict__ nb1,
                            const float* __restrict__ nb2,
                            const float* __restrict__ W0,
                            const float* __restrict__ W1,
                            const float* __restrict__ W2,
                            float* __restrict__ outN, int nRows) {
  __shared__ _Float16 wT[DIM * DIM];        // 32KB: W^T as f16, wT[n*128+k]
  __shared__ _Float16 prod[4][16 * DIM];    // 16KB: per-wave (x*nb) tile, f16
  __shared__ float    sbuf[4][3][16];       // per-wave relation scores

  const int tid  = threadIdx.x;
  const int wave = tid >> 5;
  const int lane = tid & 31;
  const int half = lane >> 4;  // K-half selector for 16-bit A/B fragments
  const int ln   = lane & 15;  // row (A) / column (B) index within tile
  const int tile = blockIdx.x * 4 + wave;
  const int row0 = tile * 16;
  const bool active = row0 < nRows;

  const float* nbs[3] = {nb0, nb1, nb2};
  const float* Ws[3]  = {W0, W1, W2};

  for (int rel = 0; rel < 3; ++rel) {
    __syncthreads();  // all waves done reading wT from previous relation
    // Stage W_rel transposed (column-major) as f16 into LDS: coalesced reads.
    const float* W = Ws[rel];
    for (int i = tid; i < DIM * DIM; i += 128) {
      const int k = i >> 7;
      const int n = i & (DIM - 1);
      wT[n * DIM + k] = (_Float16)W[i];
    }
    __syncthreads();

    if (active) {
      const float* nb = nbs[rel];
      // Stage product tile (16 x 128) as f16 into this wave's LDS buffer.
      for (int r = 0; r < 16; ++r) {
        const int rrow = min(row0 + r, nRows - 1);
        const float4 xv = *(const float4*)(x  + (size_t)rrow * DIM + lane * 4);
        const float4 nv = *(const float4*)(nb + (size_t)rrow * DIM + lane * 4);
        v4h p;
        p[0] = (_Float16)(xv.x * nv.x);
        p[1] = (_Float16)(xv.y * nv.y);
        p[2] = (_Float16)(xv.z * nv.z);
        p[3] = (_Float16)(xv.w * nv.w);
        *(v4h*)&prod[wave][r * DIM + lane * 4] = p;
      }
      // Same-wave LDS ops are in-order: safe to read prod now.

      float racc[8] = {0.f, 0.f, 0.f, 0.f, 0.f, 0.f, 0.f, 0.f};
      for (int nt = 0; nt < 8; ++nt) {       // N = 128 in 8 tiles of 16
        v8f c = {0.f, 0.f, 0.f, 0.f, 0.f, 0.f, 0.f, 0.f};
        #pragma unroll
        for (int kc = 0; kc < 4; ++kc) {     // K = 128 in 4 chunks of 32
          // A fragment: row m=ln; halves K = kc*32 + half*8 + {0..7, 16..23}
          const int abase = ln * DIM + kc * 32 + half * 8;
          const v8h a_lo = *(const v8h*)&prod[wave][abase];
          const v8h a_hi = *(const v8h*)&prod[wave][abase + 16];
          const v16h a = __builtin_shufflevector(
              a_lo, a_hi, 0, 1, 2, 3, 4, 5, 6, 7, 8, 9, 10, 11, 12, 13, 14, 15);
          // B fragment: column n = nt*16+ln; same K striping from wT.
          const int bbase = (nt * 16 + ln) * DIM + kc * 32 + half * 8;
          const v8h b_lo = *(const v8h*)&wT[bbase];
          const v8h b_hi = *(const v8h*)&wT[bbase + 16];
          const v16h b = __builtin_shufflevector(
              b_lo, b_hi, 0, 1, 2, 3, 4, 5, 6, 7, 8, 9, 10, 11, 12, 13, 14, 15);
          c = __builtin_amdgcn_wmma_f32_16x16x32_f16(
              false, a, false, b, (short)0, c, false, false);
        }
        #pragma unroll
        for (int j = 0; j < 8; ++j) {        // leaky_relu + partial rowsum
          const float h = c[j];
          racc[j] += (h >= 0.f) ? h : 0.2f * h;
        }
      }
      // Reduce across the 16 lanes holding N (stays within each lane-half).
      #pragma unroll
      for (int j = 0; j < 8; ++j) {
        float v = racc[j];
        v += __shfl_xor(v, 1);
        v += __shfl_xor(v, 2);
        v += __shfl_xor(v, 4);
        v += __shfl_xor(v, 8);
        racc[j] = v;
      }
      if (ln == 0) {  // lanes 0 and 16 hold totals for rows half*8 + j
        const float inv_sqrt_d = 0.08838834764831845f;  // 1/sqrt(128)
        #pragma unroll
        for (int j = 0; j < 8; ++j)
          sbuf[wave][rel][half * 8 + j] = racc[j] * inv_sqrt_d;
      }
    }
  }

  if (active) {
    __builtin_prefetch(nb0 + (size_t)row0 * DIM, 0, 3);
    __builtin_prefetch(nb1 + (size_t)row0 * DIM, 0, 3);
    __builtin_prefetch(nb2 + (size_t)row0 * DIM, 0, 3);
    for (int r = 0; r < 16; ++r) {
      const int rrow = row0 + r;
      if (rrow >= nRows) break;
      const float s0 = sbuf[wave][0][r];
      const float s1 = sbuf[wave][1][r];
      const float s2 = sbuf[wave][2][r];
      const float mx = fmaxf(s0, fmaxf(s1, s2));
      float e0 = __expf(s0 - mx), e1 = __expf(s1 - mx), e2 = __expf(s2 - mx);
      const float inv = 1.f / (e0 + e1 + e2);
      e0 *= inv; e1 *= inv; e2 *= inv;
      const size_t off = (size_t)rrow * DIM + lane * 4;
      const float4 a = *(const float4*)(nb0 + off);
      const float4 b = *(const float4*)(nb1 + off);
      const float4 c = *(const float4*)(nb2 + off);
      float4 o;
      o.x = a.x * e0 + b.x * e1 + c.x * e2;
      o.y = a.y * e0 + b.y * e1 + c.y * e2;
      o.z = a.z * e0 + b.z * e1 + c.z * e2;
      o.w = a.w * e0 + b.w * e1 + c.w * e2;
      *(float4*)(outN + off) = o;
    }
  }
}

extern "C" void kernel_launch(void* const* d_in, const int* in_sizes, int n_in,
                              void* d_out, int out_size, void* d_ws, size_t ws_size,
                              hipStream_t stream) {
  // setup_inputs() order:
  // 0:b 1:items_emb 2..4:src/dst/val p2p_in 5..7:p2p_out 8..10:e2p_in
  // 11..13:src/dst/val ui 14:I_p2p_in 15:I_p2p_out 16:I_e2p_in
  const float* x  = (const float*)d_in[1];
  const int n_items = in_sizes[1] / DIM;
  const int n_users = out_size / DIM - n_items;

  const int* src0 = (const int*)d_in[2];
  const int* dst0 = (const int*)d_in[3];
  const float* val0 = (const float*)d_in[4];
  const int* src1 = (const int*)d_in[5];
  const int* dst1 = (const int*)d_in[6];
  const float* val1 = (const float*)d_in[7];
  const int* src2 = (const int*)d_in[8];
  const int* dst2 = (const int*)d_in[9];
  const float* val2 = (const float*)d_in[10];
  const int* srcu = (const int*)d_in[11];
  const int* dstu = (const int*)d_in[12];
  const float* valu = (const float*)d_in[13];
  const float* W0 = (const float*)d_in[14];
  const float* W1 = (const float*)d_in[15];
  const float* W2 = (const float*)d_in[16];
  const int nE_ii = in_sizes[2];
  const int nE_ui = in_sizes[11];

  float* u_emb    = (float*)d_out;                       // [n_users, 128]
  float* neighbor = u_emb + (size_t)n_users * DIM;       // [n_items, 128]

  float* nb0 = (float*)d_ws;                             // [n_items, 128] x3
  float* nb1 = nb0 + (size_t)n_items * DIM;
  float* nb2 = nb1 + (size_t)n_items * DIM;

  // 1) zero accumulators (nb arrays in ws + u_emb slice of d_out)
  zero_kernel<<<2048, 256, 0, stream>>>(nb0, (size_t)3 * n_items * DIM / 4);
  zero_kernel<<<1024, 256, 0, stream>>>(u_emb, (size_t)n_users * DIM / 4);

  // 2) edge-parallel SpMMs (8 edges per 256-thread block, one wave per edge)
  const int blocks_ii = (nE_ii + 7) / 8;
  spmm_kernel<<<blocks_ii, 256, 0, stream>>>(src0, dst0, val0, x, nb0, nE_ii);
  spmm_kernel<<<blocks_ii, 256, 0, stream>>>(src1, dst1, val1, x, nb1, nE_ii);
  spmm_kernel<<<blocks_ii, 256, 0, stream>>>(src2, dst2, val2, x, nb2, nE_ii);
  const int blocks_ui = (nE_ui + 7) / 8;
  spmm_kernel<<<blocks_ui, 256, 0, stream>>>(srcu, dstu, valu, x, u_emb, nE_ui);

  // 3) fused gate: WMMA GEMMs + softmax + blend -> neighbor
  const int nTiles = (n_items + 15) / 16;
  const int blocks_g = (nTiles + 3) / 4;  // 4 waves (tiles) per 128-thread block
  gate_kernel<<<blocks_g, 128, 0, stream>>>(x, nb0, nb1, nb2, W0, W1, W2,
                                            neighbor, n_items);
}
